// PhysNetInteraction_80298708566685
// MI455X (gfx1250) — compile-verified, matrix operand order checked by reading, not run on hardware
//
#include <hip/hip_runtime.h>

#define DFEAT 128
#define NRBF 20

typedef __attribute__((ext_vector_type(16))) _Float16 v16h;
typedef __attribute__((ext_vector_type(8)))  float    v8f;

// Bare-metal softplus: max(x,0) + ln2 * v_log_f32(1 + v_exp_f32(-log2e*|x|)).
// exp arg <= 0 and log arg in (1,2] -> no inf/denorm cases, so the raw
// __builtin_amdgcn_{exp2f,logf} (no libm guard code) are exact enough here;
// result only feeds an f16 WMMA operand. 2 TRANS + ~4 VALU per call.
__device__ __forceinline__ float softplusf(float x) {
  float e = __builtin_amdgcn_exp2f(-1.44269504f * fabsf(x));
  return fmaxf(x, 0.f) + 0.69314718f * __builtin_amdgcn_logf(1.f + e);
}

__device__ __forceinline__ v8f wmma_f16(v16h a, v16h b, v8f c) {
  return __builtin_amdgcn_wmma_f32_16x16x32_f16(false, a, false, b, (short)0, c,
                                                false, false);
}

// A fragment (16x32 f16) from an fp32 row with softplus applied.
// Lane l holds M = l&15; element e -> k = ks*32 + ((e&8)?16:0) + ((l&16)?8:0) + (e&7)
__device__ __forceinline__ v16h make_a_sp(const float* __restrict__ row, int ks, int lane) {
  const int c8 = (lane & 16) ? 8 : 0;
  const float* b0 = row + ks * 32 + c8;
  float t[16];
  #pragma unroll
  for (int i = 0; i < 8; ++i) t[i] = b0[i];
  #pragma unroll
  for (int i = 0; i < 8; ++i) t[8 + i] = b0[16 + i];
  v16h a;
  #pragma unroll
  for (int i = 0; i < 16; ++i) a[i] = (_Float16)softplusf(t[i]);
  return a;
}

// A fragment from an LDS tile of pre-activated f16 (16 rows x 128 cols, row-major)
__device__ __forceinline__ v16h make_a_lds(const _Float16* __restrict__ sh, int ks, int lane) {
  const int c8 = (lane & 16) ? 8 : 0;
  const _Float16* b0 = sh + (lane & 15) * DFEAT + ks * 32 + c8;
  v16h a;
  #pragma unroll
  for (int i = 0; i < 8; ++i) { a[i] = b0[i]; a[8 + i] = b0[16 + i]; }
  return a;
}

// ---------------- prep kernels ----------------

__global__ void zero_f32(float* __restrict__ p, int n) {
  int t = blockIdx.x * blockDim.x + threadIdx.x;
  if (t < n) p[t] = 0.f;
}

// Pack a [128,128] f32 weight (row = output feature d, col = k) into the exact
// WMMA B-fragment order: dst[((nt*4+ks)*32+lane)*16 + e] = W[nt*16+(lane&15)][k(e,lane,ks)]
__global__ void pack_w128(const float* __restrict__ src, _Float16* __restrict__ dst) {
  int t = blockIdx.x * blockDim.x + threadIdx.x;  // 16384 threads exactly
  int e = t & 15, lane = (t >> 4) & 31, ks = (t >> 9) & 3, nt = (t >> 11) & 7;
  int d = nt * 16 + (lane & 15);
  int k = ks * 32 + ((e & 8) ? 16 : 0) + ((lane & 16) ? 8 : 0) + (e & 7);
  dst[t] = (_Float16)src[d * DFEAT + k];
}

// Pack Wg [128,20] zero-padded to K=32 into B-fragment order (single k-step).
__global__ void pack_wg(const float* __restrict__ src, _Float16* __restrict__ dst) {
  int t = blockIdx.x * blockDim.x + threadIdx.x;  // 4096 threads exactly
  int e = t & 15, lane = (t >> 4) & 31, nt = (t >> 9) & 7;
  int d = nt * 16 + (lane & 15);
  int k = ((e & 8) ? 16 : 0) + ((lane & 16) ? 8 : 0) + (e & 7);
  dst[t] = (k < NRBF) ? (_Float16)src[d * NRBF + k] : (_Float16)0.f;
}

// ---------------- edge kernel: vp scatter-add ----------------
// one wave32 per 16-edge tile

__global__ void __launch_bounds__(32) edge_kernel(
    const float* __restrict__ x, const float* __restrict__ g,
    const int* __restrict__ idx_i, const int* __restrict__ idx_j,
    const _Float16* __restrict__ WjH, const float* __restrict__ bj,
    const _Float16* __restrict__ WgH, const float* __restrict__ bg,
    float* __restrict__ v) {
  const int lane = threadIdx.x;
  const int tile = blockIdx.x * 16;
  const int nn = lane & 15;
  const int c8 = (lane & 16) ? 8 : 0;

  const int j = idx_j[tile + nn];
  const float* xr = x + (size_t)j * DFEAT;

  v8f acc[8] = {};
  #pragma unroll
  for (int ks = 0; ks < 4; ++ks) {
    v16h a = make_a_sp(xr, ks, lane);
    #pragma unroll
    for (int nt = 0; nt < 8; ++nt) {
      v16h b = *(const v16h*)(WjH + ((nt * 4 + ks) * 32 + lane) * 16);
      acc[nt] = wmma_f16(a, b, acc[nt]);
    }
  }

  // gate = g_ij @ Wg.T (K padded 20 -> 32, one WMMA per n-tile).
  // g_ij is streamed exactly once -> non-temporal so it doesn't evict L2-resident x.
  v16h ga;
  const float* gr = g + (size_t)(tile + nn) * NRBF;
  #pragma unroll
  for (int e = 0; e < 16; ++e) {
    int k = ((e & 8) ? 16 : 0) + c8 + (e & 7);
    ga[e] = (k < NRBF) ? (_Float16)__builtin_nontemporal_load(gr + k) : (_Float16)0.f;
  }
  v8f gacc[8] = {};
  #pragma unroll
  for (int nt = 0; nt < 8; ++nt) {
    v16h b = *(const v16h*)(WgH + (nt * 32 + lane) * 16);
    gacc[nt] = wmma_f16(ga, b, gacc[nt]);
  }

  // vp = softplus(acc + bj) * (gacc + bg); scatter-add into v[idx_i]
  #pragma unroll
  for (int r = 0; r < 8; ++r) {
    const int ii = idx_i[tile + r + c8];
    float* vr = v + (size_t)ii * DFEAT;
    #pragma unroll
    for (int nt = 0; nt < 8; ++nt) {
      const int n = nt * 16 + nn;
      float val = softplusf(acc[nt][r] + bj[n]) * (gacc[nt][r] + bg[n]);
      atomicAdd(vr + n, val);
    }
  }
}

// ---------------- atom kernel: v += vm ----------------

__global__ void __launch_bounds__(32) atom_kernel(
    const float* __restrict__ x, const _Float16* __restrict__ WiH,
    const float* __restrict__ bi, float* __restrict__ v) {
  const int lane = threadIdx.x;
  const int tile = blockIdx.x * 16;
  const int nn = lane & 15;
  const int c8 = (lane & 16) ? 8 : 0;
  const float* xr = x + (size_t)(tile + nn) * DFEAT;

  v8f acc[8] = {};
  #pragma unroll
  for (int ks = 0; ks < 4; ++ks) {
    v16h a = make_a_sp(xr, ks, lane);
    #pragma unroll
    for (int nt = 0; nt < 8; ++nt) {
      v16h b = *(const v16h*)(WiH + ((nt * 4 + ks) * 32 + lane) * 16);
      acc[nt] = wmma_f16(a, b, acc[nt]);
    }
  }
  #pragma unroll
  for (int r = 0; r < 8; ++r) {
    const size_t row = tile + r + c8;
    #pragma unroll
    for (int nt = 0; nt < 8; ++nt) {
      const int n = nt * 16 + nn;
      float* p = v + row * DFEAT + n;
      *p = *p + softplusf(acc[nt][r] + bi[n]);  // disjoint rows: no atomic needed
    }
  }
}

// ---------------- residual chain + final projection ----------------
// one wave32 per 16-atom tile; whole chain lives in registers, act() re-layout via LDS

__global__ void __launch_bounds__(32) resid_kernel(
    const float* __restrict__ x, const float* __restrict__ u,
    const float* __restrict__ v,
    const _Float16* __restrict__ W1H, const float* __restrict__ rb1,
    const _Float16* __restrict__ W2H, const float* __restrict__ rb2,
    const _Float16* __restrict__ WfH, const float* __restrict__ bf,
    float* __restrict__ out) {
  __shared__ _Float16 sh[16 * DFEAT];  // 4 KB, single-wave workgroup
  const int lane = threadIdx.x;
  const int tile = blockIdx.x * 16;
  const int nn = lane & 15;
  const int c8 = (lane & 16) ? 8 : 0;

  float vt[8][8];
  #pragma unroll
  for (int nt = 0; nt < 8; ++nt)
    #pragma unroll
    for (int r = 0; r < 8; ++r)
      vt[nt][r] = v[(size_t)(tile + r + c8) * DFEAT + nt * 16 + nn];

  #pragma unroll 1
  for (int l = 0; l < 3; ++l) {
    __syncthreads();
    #pragma unroll
    for (int nt = 0; nt < 8; ++nt)
      #pragma unroll
      for (int r = 0; r < 8; ++r)
        sh[(r + c8) * DFEAT + nt * 16 + nn] = (_Float16)softplusf(vt[nt][r]);
    __syncthreads();

    v8f h[8] = {};
    #pragma unroll
    for (int ks = 0; ks < 4; ++ks) {
      v16h a = make_a_lds(sh, ks, lane);
      #pragma unroll
      for (int nt = 0; nt < 8; ++nt) {
        v16h b = *(const v16h*)(W1H + l * 16384 + ((nt * 4 + ks) * 32 + lane) * 16);
        h[nt] = wmma_f16(a, b, h[nt]);
      }
    }

    __syncthreads();
    #pragma unroll
    for (int nt = 0; nt < 8; ++nt)
      #pragma unroll
      for (int r = 0; r < 8; ++r)
        sh[(r + c8) * DFEAT + nt * 16 + nn] =
            (_Float16)softplusf(h[nt][r] + rb1[l * DFEAT + nt * 16 + nn]);
    __syncthreads();

    v8f h2[8] = {};
    #pragma unroll
    for (int ks = 0; ks < 4; ++ks) {
      v16h a = make_a_lds(sh, ks, lane);
      #pragma unroll
      for (int nt = 0; nt < 8; ++nt) {
        v16h b = *(const v16h*)(W2H + l * 16384 + ((nt * 4 + ks) * 32 + lane) * 16);
        h2[nt] = wmma_f16(a, b, h2[nt]);
      }
    }
    #pragma unroll
    for (int nt = 0; nt < 8; ++nt)
      #pragma unroll
      for (int r = 0; r < 8; ++r)
        vt[nt][r] += h2[nt][r] + rb2[l * DFEAT + nt * 16 + nn];
  }

  // out = u*x + softplus(v) @ Wf.T + bf
  __syncthreads();
  #pragma unroll
  for (int nt = 0; nt < 8; ++nt)
    #pragma unroll
    for (int r = 0; r < 8; ++r)
      sh[(r + c8) * DFEAT + nt * 16 + nn] = (_Float16)softplusf(vt[nt][r]);
  __syncthreads();

  v8f f[8] = {};
  #pragma unroll
  for (int ks = 0; ks < 4; ++ks) {
    v16h a = make_a_lds(sh, ks, lane);
    #pragma unroll
    for (int nt = 0; nt < 8; ++nt) {
      v16h b = *(const v16h*)(WfH + ((nt * 4 + ks) * 32 + lane) * 16);
      f[nt] = wmma_f16(a, b, f[nt]);
    }
  }
  #pragma unroll
  for (int r = 0; r < 8; ++r) {
    const size_t row = tile + r + c8;
    #pragma unroll
    for (int nt = 0; nt < 8; ++nt) {
      const int n = nt * 16 + nn;
      float o = u[n] * x[row * DFEAT + n] + f[nt][r] + bf[n];
      __builtin_nontemporal_store(o, out + row * DFEAT + n);  // written once, never re-read
    }
  }
}

// ---------------- launch ----------------

extern "C" void kernel_launch(void* const* d_in, const int* in_sizes, int n_in,
                              void* d_out, int out_size, void* d_ws, size_t ws_size,
                              hipStream_t stream) {
  const float* x    = (const float*)d_in[0];
  const float* g    = (const float*)d_in[1];
  const int*   iidx = (const int*)d_in[2];
  const int*   jidx = (const int*)d_in[3];
  // d_in[4] = n_atoms scalar (unused; derived from in_sizes)
  const float* u    = (const float*)d_in[5];
  const float* Wf   = (const float*)d_in[6];
  const float* bf   = (const float*)d_in[7];
  const float* Wg   = (const float*)d_in[8];
  const float* bg   = (const float*)d_in[9];
  const float* Wj   = (const float*)d_in[10];
  const float* bj   = (const float*)d_in[11];
  const float* Wi   = (const float*)d_in[12];
  const float* bi   = (const float*)d_in[13];
  const float* rw1  = (const float*)d_in[14];
  const float* rb1  = (const float*)d_in[15];
  const float* rw2  = (const float*)d_in[16];
  const float* rb2  = (const float*)d_in[17];
  float* out = (float*)d_out;

  const int N = in_sizes[0] / DFEAT;   // 50000 (multiple of 16)
  const int E = in_sizes[2];           // 640000 (multiple of 16)

  char* ws = (char*)d_ws;
  float* v = (float*)ws;                                   // N*128 f32 accumulator
  _Float16* pw  = (_Float16*)(ws + (size_t)N * DFEAT * sizeof(float));
  _Float16* WjH = pw;                  // 16384 halves each
  _Float16* WiH = pw + 16384;
  _Float16* WfH = pw + 2 * 16384;
  _Float16* W1H = pw + 3 * 16384;      // 3 layers consecutive
  _Float16* W2H = pw + 6 * 16384;
  _Float16* WgH = pw + 9 * 16384;      // 4096 halves

  // prep: zero accumulator, pack weights to f16 B-fragment layout
  zero_f32<<<(N * DFEAT + 255) / 256, 256, 0, stream>>>(v, N * DFEAT);
  pack_w128<<<64, 256, 0, stream>>>(Wj, WjH);
  pack_w128<<<64, 256, 0, stream>>>(Wi, WiH);
  pack_w128<<<64, 256, 0, stream>>>(Wf, WfH);
  for (int l = 0; l < 3; ++l) {
    pack_w128<<<64, 256, 0, stream>>>(rw1 + (size_t)l * 16384, W1H + (size_t)l * 16384);
    pack_w128<<<64, 256, 0, stream>>>(rw2 + (size_t)l * 16384, W2H + (size_t)l * 16384);
  }
  pack_wg<<<16, 256, 0, stream>>>(Wg, WgH);

  // main pipeline (stream-ordered)
  edge_kernel<<<E / 16, 32, 0, stream>>>(x, g, iidx, jidx, WjH, bj, WgH, bg, v);
  atom_kernel<<<N / 16, 32, 0, stream>>>(x, WiH, bi, v);
  resid_kernel<<<N / 16, 32, 0, stream>>>(x, u, v, W1H, rb1, W2H, rb2, WfH, bf, out);
}
